// gcs_layer_44341242364284
// MI455X (gfx1250) — matrix operand order repaired and checked by reading, not compiled, not run on hardware
//
#include <hip/hip_runtime.h>
#include <hip/hip_bf16.h>
#include <math.h>

// ---------------------------------------------------------------------------
// Types for CDNA5 WMMA (wave32): v16bf = 16 bf16 (8 VGPRs), v8f = 8 f32.
// ---------------------------------------------------------------------------
typedef __attribute__((ext_vector_type(16))) __bf16 v16bf;
typedef __attribute__((ext_vector_type(8)))  float  v8f;
typedef __attribute__((ext_vector_type(4)))  int    v4i;

union Frag16 { v16bf v; uint4 u[2]; };

#define AS1 __attribute__((address_space(1)))
#define AS3 __attribute__((address_space(3)))

#if __has_builtin(__builtin_amdgcn_global_load_async_to_lds_b128) && \
    __has_builtin(__builtin_amdgcn_s_wait_asynccnt)
#define HAVE_ASYNC_LDS 1
#else
#define HAVE_ASYNC_LDS 0
#endif

__device__ __forceinline__ unsigned short f2bf(float f) {
  union { float f; unsigned u; } c; c.f = f;
  unsigned u = c.u + 0x7FFFu + ((c.u >> 16) & 1u);   // round-to-nearest-even
  return (unsigned short)(u >> 16);
}
__device__ __forceinline__ float bflo(unsigned u) { return __uint_as_float(u << 16); }
__device__ __forceinline__ float bfhi(unsigned u) { return __uint_as_float(u & 0xFFFF0000u); }

__device__ __forceinline__ void atomicMaxFloat(float* addr, float val) {
  // Sign-aware trick: works with init = -inf (0xFF800000).
  if (val >= 0.0f) atomicMax((int*)addr, __float_as_int(val));
  else             atomicMin((unsigned int*)addr, __float_as_uint(val));
}

// 16-byte copy global -> LDS. Async (ASYNCcnt / gfx1250 path) when available.
__device__ __forceinline__ void cp16(unsigned short* ldst, const unsigned short* gsrc) {
#if HAVE_ASYNC_LDS
  __builtin_amdgcn_global_load_async_to_lds_b128((AS1 v4i*)gsrc, (AS3 v4i*)ldst, 0, 0);
#else
  *(uint4*)ldst = *(const uint4*)gsrc;
#endif
}
__device__ __forceinline__ void wait_async_copies() {
#if HAVE_ASYNC_LDS
  __builtin_amdgcn_s_wait_asynccnt(0);
#endif
}

#define DIMC 256
#define HEADS_SCALE (1.0f / 2048.0f)   // 1/(8*NUM_HEADS*HEAD_DIM) = 1/(8*4*64)

// ---------------------------------------------------------------------------
// 0) init: zero output accumulator, node sums, global stats; node_max = -inf
// ---------------------------------------------------------------------------
__global__ void k_init(float* __restrict__ out, float* __restrict__ node_max,
                       float* __restrict__ node_sum, double* __restrict__ gstat,
                       int n_out, int n_nodes) {
  int i = blockIdx.x * blockDim.x + threadIdx.x;
  int stride = gridDim.x * blockDim.x;
  for (int t = i; t < n_out; t += stride) {
    out[t] = 0.0f;
    if (t < n_nodes) { node_max[t] = -INFINITY; node_sum[t] = 0.0f; }
  }
  if (i == 0) { gstat[0] = 0.0; gstat[1] = 0.0; }
}

// ---------------------------------------------------------------------------
// 1) convert h (f32) -> bf16 row-major
// ---------------------------------------------------------------------------
__global__ void k_cvt_h(const float* __restrict__ h, unsigned short* __restrict__ hbf, int n) {
  int i = blockIdx.x * blockDim.x + threadIdx.x;
  int stride = gridDim.x * blockDim.x;
  for (int t = i; t < n; t += stride) hbf[t] = f2bf(h[t]);
}

// ---------------------------------------------------------------------------
// 2) pack the three 256x256 weight matrices into WMMA B-fragment layout.
//    B0[k][n] = W[n][k] (for v = h @ W^T), B1 = attn_q, B2 = attn_k.
//    Fragment (kt, nt) = 32 lanes x 16 bf16, contiguous per lane; the 16
//    fragments of one kt are a contiguous 16KB chunk (async-copied to LDS).
//    Lane l<16 element j: K = 32*kt + (j&7) + 16*(j>>3), N = 16*nt + l
//    Lane l>=16:         K += 8
// ---------------------------------------------------------------------------
__global__ void k_pack_b(const float* __restrict__ W, const float* __restrict__ Aq,
                         const float* __restrict__ Ak, unsigned short* __restrict__ packB) {
  int idx = blockIdx.x * blockDim.x + threadIdx.x;   // mat*4096 + frag*32 + lane
  if (idx >= 3 * 128 * 32) return;
  int lane = idx & 31;
  int frag = (idx >> 5) & 127;
  int mat  = idx >> 12;
  int kt = frag >> 4, ntile = frag & 15;
  unsigned short* dst = packB + ((size_t)idx) * 16;
  int khalf = (lane >> 4) << 4;
  int n = ntile * 16 + (lane & 15);
#pragma unroll
  for (int j = 0; j < 16; ++j) {
    int kk = kt * 32 + (khalf >> 1) + (j & 7) + ((j >> 3) << 4);
    float val;
    if (mat == 0)      val = W [(size_t)n  * DIMC + kk];
    else if (mat == 1) val = Aq[(size_t)kk * DIMC + n];
    else               val = Ak[(size_t)kk * DIMC + n];
    dst[j] = f2bf(val);
  }
}

// ---------------------------------------------------------------------------
// 3) fused GEMM: blockIdx.y selects output {0: v (f32, +bias), 1: q (bf16),
//    2: k (bf16)}. Block = 8 waves = 64(M) x 256(N); each wave does 16x128
//    with 8 v8f accumulators. B tiles (16KB per K-step) are double-buffered
//    in LDS via async global->LDS copies; A fragments and LDS B fragments
//    are software-pipelined in registers so WMMAs overlap the fetches.
// ---------------------------------------------------------------------------
__global__ void __launch_bounds__(256)
k_gemm(const unsigned short* __restrict__ hbf, const unsigned short* __restrict__ packB,
       const float* __restrict__ bvec, float* __restrict__ vout,
       unsigned short* __restrict__ qbf, unsigned short* __restrict__ kbf, int nrows) {
  __shared__ __align__(32) unsigned short smem[2][16 * 32 * 16];  // 2 x 16KB
  const int mat   = blockIdx.y;
  const int tid   = threadIdx.x;
  const int wave  = tid >> 5;              // 0..7
  const int lane  = tid & 31;
  const int mstrip = wave >> 1;            // 0..3
  const int nhalf  = wave & 1;             // 0..1 (cols 0-127 / 128-255)
  const int l16    = lane & 15;
  const int hi     = lane >> 4;            // lane half selects K sub-range

  const int m0 = blockIdx.x * 64 + mstrip * 16;
  int arow = m0 + l16;
  if (arow >= nrows) arow = nrows - 1;     // clamp: EXEC must stay all-ones
  const unsigned short* Abase = hbf + (size_t)arow * DIMC;
  const unsigned short* Bbase = packB + (size_t)mat * (128 * 32 * 16);

  v8f acc[8];
#pragma unroll
  for (int i = 0; i < 8; ++i) acc[i] = v8f{};

  // Stage B tile for kt=0 (16KB contiguous; 64B per thread).
  {
    const unsigned short* s = Bbase + tid * 32;
    unsigned short* d = &smem[0][tid * 32];
#pragma unroll
    for (int j = 0; j < 4; ++j) cp16(d + j * 8, s + j * 8);
  }

  // A fragment for kt=0 (lane<16: K {0..7,16..23}; lane>=16: K {8..15,24..31}).
  Frag16 a_cur, a_nxt;
  a_cur.u[0] = *(const uint4*)(Abase + hi * 8);
  a_cur.u[1] = *(const uint4*)(Abase + hi * 8 + 16);

  for (int kt = 0; kt < 8; ++kt) {
    wait_async_copies();      // this wave's copies for buffer kt&1 landed
    __syncthreads();          // all waves' copies landed; prev buffer free

    if (kt < 7) {
      // prefetch next A fragment
      const int kb = (kt + 1) * 32 + hi * 8;
      a_nxt.u[0] = *(const uint4*)(Abase + kb);
      a_nxt.u[1] = *(const uint4*)(Abase + kb + 16);
      // async-stage next B tile into the other LDS buffer
      const unsigned short* s = Bbase + (kt + 1) * 8192 + tid * 32;
      unsigned short* d = &smem[(kt + 1) & 1][tid * 32];
#pragma unroll
      for (int j = 0; j < 4; ++j) cp16(d + j * 8, s + j * 8);
    }

    const unsigned short* bufp = &smem[kt & 1][0];
    Frag16 b0, b1;
    b0 = *(const Frag16*)(bufp + ((size_t)(nhalf * 8) * 32 + lane) * 16);
#pragma unroll
    for (int nt = 0; nt < 8; ++nt) {
      if (nt < 7)
        b1 = *(const Frag16*)(bufp + ((size_t)(nhalf * 8 + nt + 1) * 32 + lane) * 16);
      acc[nt] = __builtin_amdgcn_wmma_f32_16x16x32_bf16(
          false, a_cur.v, false, b0.v, (short)0, acc[nt], false, false);
      b0 = b1;
    }
    a_cur = a_nxt;
  }

  // Epilogue. C/D layout: VGPR i, lanes 0-15 -> M=i, N=lane; lanes 16-31 -> M=8+i.
#pragma unroll
  for (int nt = 0; nt < 8; ++nt) {
    const int col = nhalf * 128 + nt * 16 + l16;
#pragma unroll
    for (int i = 0; i < 8; ++i) {
      const int row = m0 + hi * 8 + i;
      if (row < nrows) {
        const float val = acc[nt][i];
        if (mat == 0)      vout[(size_t)row * DIMC + col] = val + bvec[col];
        else if (mat == 1) qbf [(size_t)row * DIMC + col] = f2bf(val);
        else               kbf [(size_t)row * DIMC + col] = f2bf(val);
      }
    }
  }
}

// ---------------------------------------------------------------------------
// 4) per-edge raw score: a[e] = dot(q[src], k[dst]) / 2048 ; accumulate
//    global sum / sumsq (block-reduced, double atomics). One wave per edge;
//    q/k rows are bf16 and L2-resident (2 x 25.6MB << 192MB L2).
// ---------------------------------------------------------------------------
__global__ void __launch_bounds__(256)
k_edge_score(const unsigned short* __restrict__ qbf, const unsigned short* __restrict__ kbf,
             const int* __restrict__ src, const int* __restrict__ dst,
             float* __restrict__ a_raw, double* __restrict__ gstat, int nedges) {
  const int wid  = threadIdx.x >> 5;
  const int lane = threadIdx.x & 31;
  const int e = blockIdx.x * 8 + wid;
  __shared__ float part[8];
  float aval = 0.0f;
  if (e < nedges) {
    const int s = src[e], d = dst[e];
    const uint4 qa = *(const uint4*)(qbf + (size_t)s * DIMC + lane * 8);
    const uint4 ka = *(const uint4*)(kbf + (size_t)d * DIMC + lane * 8);
    const unsigned* qu = (const unsigned*)&qa;
    const unsigned* ku = (const unsigned*)&ka;
    float sum = 0.0f;
#pragma unroll
    for (int i = 0; i < 4; ++i)
      sum += bflo(qu[i]) * bflo(ku[i]) + bfhi(qu[i]) * bfhi(ku[i]);
#pragma unroll
    for (int off = 16; off >= 1; off >>= 1) sum += __shfl_xor(sum, off, 32);
    aval = sum * HEADS_SCALE;
    if (lane == 0) a_raw[e] = aval;
  }
  if (lane == 0) part[wid] = (e < nedges) ? aval : 0.0f;
  __syncthreads();
  if (threadIdx.x == 0) {
    double s1 = 0.0, s2 = 0.0;
#pragma unroll
    for (int i = 0; i < 8; ++i) { double x = part[i]; s1 += x; s2 += x * x; }
    atomicAdd(&gstat[0], s1);
    atomicAdd(&gstat[1], s2);
  }
}

// ---------------------------------------------------------------------------
// 5) finalize mean / inv(std*T)
// ---------------------------------------------------------------------------
__global__ void k_stats(const double* __restrict__ gstat, float* __restrict__ params, int nedges) {
  if (threadIdx.x == 0 && blockIdx.x == 0) {
    double mean = gstat[0] / (double)nedges;
    double var  = gstat[1] / (double)nedges - mean * mean;
    if (var < 0.0) var = 0.0;
    params[0] = (float)mean;
    params[1] = (float)(1.0 / (sqrt(var) * 0.5));   // /std /TEMPERATURE
  }
}

// ---------------------------------------------------------------------------
// 6) standardize + segment max over dst
// ---------------------------------------------------------------------------
__global__ void k_std_max(float* __restrict__ a_raw, const int* __restrict__ dst,
                          const float* __restrict__ params, float* __restrict__ node_max,
                          int nedges) {
  const float mean = params[0], inv = params[1];
  int i = blockIdx.x * blockDim.x + threadIdx.x;
  int stride = gridDim.x * blockDim.x;
  for (int e = i; e < nedges; e += stride) {
    float a = (a_raw[e] - mean) * inv;
    a_raw[e] = a;
    atomicMaxFloat(&node_max[dst[e]], a);
  }
}

// ---------------------------------------------------------------------------
// 7) exp + segment sum over dst
// ---------------------------------------------------------------------------
__global__ void k_exp_sum(const float* __restrict__ a_raw, const int* __restrict__ dst,
                          const float* __restrict__ node_max, float* __restrict__ e_buf,
                          float* __restrict__ node_sum, int nedges) {
  int i = blockIdx.x * blockDim.x + threadIdx.x;
  int stride = gridDim.x * blockDim.x;
  for (int e = i; e < nedges; e += stride) {
    int d = dst[e];
    float ev = expf(a_raw[e] - node_max[d]);
    e_buf[e] = ev;
    atomicAdd(&node_sum[d], ev);
  }
}

// ---------------------------------------------------------------------------
// 8) normalize + scatter message pass: out[dst] += a_e * v[src]; also emit a_e.
//    One wave per edge, 8 floats per lane, float atomics into out rows.
// ---------------------------------------------------------------------------
__global__ void __launch_bounds__(256)
k_scatter(const float* __restrict__ e_buf, const float* __restrict__ node_sum,
          const int* __restrict__ src, const int* __restrict__ dst,
          const float* __restrict__ v, float* __restrict__ out,
          float* __restrict__ a_out, int nedges) {
  const int e = blockIdx.x * 8 + (threadIdx.x >> 5);
  if (e >= nedges) return;
  const int lane = threadIdx.x & 31;
  const int s = src[e], d = dst[e];
  const float af = e_buf[e] / node_sum[d];
  if (lane == 0) a_out[e] = af;
  const float* vp = v + (size_t)s * DIMC + lane * 8;
  const float4 v0 = *(const float4*)(vp);
  const float4 v1 = *(const float4*)(vp + 4);
  float* op = out + (size_t)d * DIMC + lane * 8;
  atomicAdd(op + 0, af * v0.x); atomicAdd(op + 1, af * v0.y);
  atomicAdd(op + 2, af * v0.z); atomicAdd(op + 3, af * v0.w);
  atomicAdd(op + 4, af * v1.x); atomicAdd(op + 5, af * v1.y);
  atomicAdd(op + 6, af * v1.z); atomicAdd(op + 7, af * v1.w);
}

// ---------------------------------------------------------------------------
// launch
// ---------------------------------------------------------------------------
extern "C" void kernel_launch(void* const* d_in, const int* in_sizes, int n_in,
                              void* d_out, int out_size, void* d_ws, size_t ws_size,
                              hipStream_t stream) {
  const float* h  = (const float*)d_in[0];
  const int* src  = (const int*)d_in[1];
  const int* dst  = (const int*)d_in[2];
  const float* Aq = (const float*)d_in[3];
  const float* Ak = (const float*)d_in[4];
  const float* W  = (const float*)d_in[5];
  const float* bv = (const float*)d_in[6];

  const int N = in_sizes[0] / DIMC;      // 50000
  const int E = in_sizes[1];             // 800000

  // ws carve-up (256B aligned)
  char* base = (char*)d_ws;
  size_t cur = 0;
  auto take = [&](size_t bytes) -> char* {
    size_t o = (cur + 255) & ~(size_t)255; cur = o + bytes; return base + o;
  };
  unsigned short* hbf   = (unsigned short*)take((size_t)N * DIMC * 2);
  unsigned short* packB = (unsigned short*)take((size_t)3 * 128 * 32 * 16 * 2);
  float*          vbuf  = (float*)take((size_t)N * DIMC * 4);
  unsigned short* qbf   = (unsigned short*)take((size_t)N * DIMC * 2);
  unsigned short* kbf   = (unsigned short*)take((size_t)N * DIMC * 2);
  float*          a_raw = (float*)take((size_t)E * 4);
  float*          e_buf = (float*)take((size_t)E * 4);
  float*          nmax  = (float*)take((size_t)N * 4);
  float*          nsum  = (float*)take((size_t)N * 4);
  double*         gstat = (double*)take(16);
  float*          prm   = (float*)take(8);

  float* out_nodes = (float*)d_out;                // [N, 256]
  float* out_a     = out_nodes + (size_t)N * DIMC; // [E]

  const int nElems = N * DIMC;

  k_init<<<4096, 256, 0, stream>>>(out_nodes, nmax, nsum, gstat, nElems, N);
  k_cvt_h<<<4096, 256, 0, stream>>>(h, hbf, nElems);
  k_pack_b<<<(3 * 128 * 32 + 255) / 256, 256, 0, stream>>>(W, Aq, Ak, packB);

  dim3 gg((N + 63) / 64, 3);
  k_gemm<<<gg, 256, 0, stream>>>(hbf, packB, bv, vbuf, qbf, kbf, N);

  k_edge_score<<<(E + 7) / 8, 256, 0, stream>>>(qbf, kbf, src, dst, a_raw, gstat, E);
  k_stats<<<1, 32, 0, stream>>>(gstat, prm, E);
  k_std_max<<<2048, 256, 0, stream>>>(a_raw, dst, prm, nmax, E);
  k_exp_sum<<<2048, 256, 0, stream>>>(a_raw, dst, nmax, e_buf, nsum, E);
  k_scatter<<<(E + 7) / 8, 256, 0, stream>>>(e_buf, nsum, src, dst, vbuf, out_nodes, out_a, E);
}
